// GeometricProjector_76493367542217
// MI455X (gfx1250) — matrix operand order repaired and checked by reading, not compiled, not run on hardware
//
#include <hip/hip_runtime.h>

// ---------------- problem constants ----------------
#define Bq   64
#define Tq   2048
#define CIN  44
#define Dq   512
#define DCq  640           // combined dim
#define REDq 160
#define EPSq 1e-5f
#define BT   (Bq * Tq)     // 131072 tokens

typedef __attribute__((ext_vector_type(16))) __bf16 v16bf;
typedef __attribute__((ext_vector_type(8)))  __bf16 v8bf;
typedef __attribute__((ext_vector_type(8)))  float  v8f;

// ---------------- workspace layout (bytes) ----------------
static constexpr size_t alignup(size_t x) { return (x + 255) & ~size_t(255); }
static constexpr size_t OFF_W1P  = 0;                                          // 2*40*512 bf16
static constexpr size_t OFF_W2P  = alignup(OFF_W1P  + (size_t)2*40*512*2);     // 20*40*512 bf16
static constexpr size_t OFF_GWP  = alignup(OFF_W2P  + (size_t)20*40*512*2);
static constexpr size_t OFF_PWP  = alignup(OFF_GWP  + (size_t)20*40*512*2);
static constexpr size_t OFF_OWP  = alignup(OFF_PWP  + (size_t)20*40*512*2);    // 20*32*512 bf16
static constexpr size_t OFF_B1   = alignup(OFF_OWP  + (size_t)20*32*512*2);    // 640 f32
static constexpr size_t OFF_B2   = alignup(OFF_B1   + 640*4);                  // 640 f32
static constexpr size_t OFF_STAT = alignup(OFF_B2   + 640*4);                  // zsum 40960 + gnsum 512 + gnsq 512 f32
static constexpr size_t OFF_SE   = alignup(OFF_STAT + 41984*4);                // 64*640 f32
static constexpr size_t OFF_ZBUF = alignup(OFF_SE   + (size_t)Bq*DCq*4);       // BT*DCq bf16
static constexpr size_t OFF_HBUF = alignup(OFF_ZBUF + (size_t)BT*DCq*2);       // BT*Dq f32

// ---------------- fast math (avoid IEEE div/sqrt chains) ----------------
__device__ __forceinline__ float fast_sigmoid(float t) {
    return __builtin_amdgcn_rcpf(1.f + __expf(-t));       // v_rcp_f32 + v_exp_f32
}
__device__ __forceinline__ float fast_rsqrt(float x) {
    return __builtin_amdgcn_rsqf(x);                      // v_rsq_f32
}

// ---------------- WMMA fragment helpers ----------------
// A-matrix 16x32 bf16: lane m=l&15, half=l>>4; elements e=0..15
// hold K = 16*half + e (matches packed B)  -> two 16B LDS loads.
__device__ __forceinline__ v16bf load_a(const __bf16* smem, int ldk, int kBase, int lane) {
    int m = lane & 15, half = lane >> 4;
    const __bf16* p = smem + m * ldk + kBase + 16 * half;
    v8bf lo = *(const v8bf*)p;
    v8bf hi = *(const v8bf*)(p + 8);
    v16bf a;
#pragma unroll
    for (int e = 0; e < 8; ++e) { a[e] = lo[e]; a[e + 8] = hi[e]; }
    return a;
}

// B-matrix 32x16 bf16 packed in fragment order: tile holds 512 bf16,
// lane's 16 values contiguous (32B) -> straight vector load from global/L2.
__device__ __forceinline__ v16bf load_b(const __bf16* tileBase, int lane) {
    return *(const v16bf*)(tileBase + lane * 16);
}

__device__ __forceinline__ v8f bwmma(v16bf a, v16bf b, v8f c) {
    return __builtin_amdgcn_wmma_f32_16x16x32_bf16(false, a, false, b, (short)0, c, false, false);
}

__device__ __forceinline__ float wred(float v) {
#pragma unroll
    for (int off = 16; off > 0; off >>= 1) v += __shfl_xor(v, off);
    return v;
}

// ---------------- prep kernels ----------------
__global__ __launch_bounds__(256) void k_zero(float* p, int n) {
    int i = blockIdx.x * 256 + threadIdx.x;
    if (i < n) p[i] = 0.f;
}

// generic pack: w[K][N] row-major fp32 -> bf16 fragment tiles (Ktiles x Ntiles x 512)
__global__ __launch_bounds__(256) void k_pack(const float* __restrict__ w, int Kreal, int N,
                                              int Ktiles, __bf16* __restrict__ dst) {
    int i = blockIdx.x * 256 + threadIdx.x;
    int Ntiles = N >> 4;
    int total = Ktiles * Ntiles * 512;
    if (i >= total) return;
    int tile = i >> 9, within = i & 511, lane = within >> 4, e = within & 15;
    int kt = tile / Ntiles, nt = tile - kt * Ntiles;
    int n = nt * 16 + (lane & 15);
    int k = kt * 32 + (lane >> 4) * 16 + e;
    float v = (k < Kreal) ? w[(size_t)k * N + n] : 0.f;
    dst[i] = (__bf16)v;
}

// block-diagonal concat of layer-1 weights: [44(->64 pad)] x [640]
__global__ __launch_bounds__(256) void k_pack_w1(const float* __restrict__ hw1,
                                                 const float* __restrict__ lw1,
                                                 const float* __restrict__ ew1,
                                                 __bf16* __restrict__ dst) {
    int i = blockIdx.x * 256 + threadIdx.x;
    const int Ntiles = 40, total = 2 * Ntiles * 512;
    if (i >= total) return;
    int tile = i >> 9, within = i & 511, lane = within >> 4, e = within & 15;
    int kt = tile / Ntiles, nt = tile - kt * Ntiles;
    int n = nt * 16 + (lane & 15);
    int k = kt * 32 + (lane >> 4) * 16 + e;
    float v = 0.f;
    if (k < 14 && n < 256)                             v = hw1[k * 256 + n];
    else if (k >= 14 && k < 36 && n >= 256 && n < 512) v = lw1[(k - 14) * 256 + (n - 256)];
    else if (k >= 36 && k < 44 && n >= 512)            v = ew1[(k - 36) * 128 + (n - 512)];
    dst[i] = (__bf16)v;
}

// block-diagonal concat of layer-2 weights: [640] x [640]
__global__ __launch_bounds__(256) void k_pack_w2(const float* __restrict__ hw2,
                                                 const float* __restrict__ lw2,
                                                 const float* __restrict__ ew2,
                                                 __bf16* __restrict__ dst) {
    int i = blockIdx.x * 256 + threadIdx.x;
    const int Ntiles = 40, total = 20 * Ntiles * 512;
    if (i >= total) return;
    int tile = i >> 9, within = i & 511, lane = within >> 4, e = within & 15;
    int kt = tile / Ntiles, nt = tile - kt * Ntiles;
    int n = nt * 16 + (lane & 15);
    int k = kt * 32 + (lane >> 4) * 16 + e;
    float v = 0.f;
    if (k < 256 && n < 256)                              v = hw2[k * 256 + n];
    else if (k >= 256 && k < 512 && n >= 256 && n < 512) v = lw2[(k - 256) * 256 + (n - 256)];
    else if (k >= 512 && n >= 512)                       v = ew2[(k - 512) * 128 + (n - 512)];
    dst[i] = (__bf16)v;
}

__global__ __launch_bounds__(256) void k_pack_bias(
    const float* hb1, const float* lb1, const float* eb1,
    const float* hb2, const float* lb2, const float* eb2,
    float* b1, float* b2) {
    int i = blockIdx.x * 256 + threadIdx.x;
    if (i >= DCq) return;
    b1[i] = (i < 256) ? hb1[i] : (i < 512) ? lb1[i - 256] : eb1[i - 512];
    b2[i] = (i < 256) ? hb2[i] : (i < 512) ? lb2[i - 256] : eb2[i - 512];
}

// ---------------- K1: grouped projections -> z (bf16) + column sums ----------------
// M = 32 tokens per block: two 16-row A tiles share every B fragment.
__global__ __launch_bounds__(256) void k_proj(
    const float* __restrict__ x, const __bf16* __restrict__ w1p,
    const __bf16* __restrict__ w2p, const float* __restrict__ b1,
    const float* __restrict__ b2, __bf16* __restrict__ zbuf,
    float* __restrict__ zsum) {
    __shared__ __bf16 xs[32 * 64];    // 32 tokens x 64 (44 real + zero pad)
    __shared__ __bf16 us[32 * DCq];   // layer-1 activations (silu), bf16
    int tid = threadIdx.x;
    long long t0 = (long long)blockIdx.x * 32;
    int b = (int)(t0 / Tq);

    for (int i = tid; i < 32 * 64; i += 256) {
        int m = i >> 6, c = i & 63;
        float v = (c < CIN) ? x[(t0 + m) * CIN + c] : 0.f;
        xs[i] = (__bf16)v;
    }
    __syncthreads();

    int wv = tid >> 5, lane = tid & 31, nc = lane & 15, half = lane >> 4;

    // layer 1: K = 64 (padded), N = 640 -> 40 tiles, 5 per wave
    for (int nt = wv; nt < 40; nt += 8) {
        v8f c0 = {}, c1 = {};
#pragma unroll
        for (int kt = 0; kt < 2; ++kt) {
            v16bf bb = load_b(w1p + ((size_t)(kt * 40 + nt)) * 512, lane);
            v16bf a0 = load_a(xs,           64, kt * 32, lane);
            v16bf a1 = load_a(xs + 16 * 64, 64, kt * 32, lane);
            c0 = bwmma(a0, bb, c0);
            c1 = bwmma(a1, bb, c1);
        }
        int n = nt * 16 + nc;
        float bias = b1[n];
#pragma unroll
        for (int v = 0; v < 8; ++v) {
            int m = v + 8 * half;
            float ta = c0[v] + bias;
            float tb = c1[v] + bias;
            us[m * DCq + n]        = (__bf16)(ta * fast_sigmoid(ta));   // silu
            us[(m + 16) * DCq + n] = (__bf16)(tb * fast_sigmoid(tb));
        }
    }
    __syncthreads();

    // layer 2: block-diagonal 640x640; restrict K to the tile's block
    for (int nt = wv; nt < 40; nt += 8) {
        int k0 = (nt < 16) ? 0 : (nt < 32) ? 8 : 16;
        int nk = (nt < 32) ? 8 : 4;
        v8f c0 = {}, c1 = {};
        for (int kk = 0; kk < nk; ++kk) {
            int kt = k0 + kk;
            v16bf bb = load_b(w2p + ((size_t)(kt * 40 + nt)) * 512, lane);
            v16bf a0 = load_a(us,            DCq, kt * 32, lane);
            v16bf a1 = load_a(us + 16 * DCq, DCq, kt * 32, lane);
            c0 = bwmma(a0, bb, c0);
            c1 = bwmma(a1, bb, c1);
        }
        int n = nt * 16 + nc;
        float bias = b2[n];
        float colsum = 0.f;
#pragma unroll
        for (int v = 0; v < 8; ++v) {
            int m = v + 8 * half;
            float ta = c0[v] + bias;
            float tb = c1[v] + bias;
            zbuf[(t0 + m) * DCq + n]      = (__bf16)ta;
            zbuf[(t0 + m + 16) * DCq + n] = (__bf16)tb;
            colsum += ta + tb;
        }
        colsum += __shfl_xor(colsum, 16);          // combine the two 8-row halves
        if (half == 0) atomicAdd(&zsum[b * DCq + n], colsum);
    }
}

// ---------------- K2: SE block (tiny) ----------------
__global__ __launch_bounds__(256) void k_se(
    const float* __restrict__ zsum, const float* __restrict__ w1,
    const float* __restrict__ w2, float* __restrict__ se) {
    __shared__ float y[DCq];
    __shared__ float r[REDq];
    int b = blockIdx.x, tid = threadIdx.x;
    for (int i = tid; i < DCq; i += 256) y[i] = zsum[b * DCq + i] * (1.f / Tq);
    __syncthreads();
    for (int j = tid; j < REDq; j += 256) {
        float acc = 0.f;
        for (int k = 0; k < DCq; ++k) acc += y[k] * w1[k * REDq + j];
        r[j] = acc > 0.f ? acc : 0.f;
    }
    __syncthreads();
    for (int n = tid; n < DCq; n += 256) {
        float acc = 0.f;
        for (int k = 0; k < REDq; ++k) acc += r[k] * w2[k * DCq + n];
        se[b * DCq + n] = fast_sigmoid(acc);
    }
}

// ---------------- K3: SE-scale, GLU gate, LayerNorm, output proj, GN stats ----------------
// M = 32 tokens per block; 80 KB LDS (zt + hb, both bf16); hn aliases zt.
__global__ __launch_bounds__(256) void k_gate(
    const __bf16* __restrict__ zbuf, const float* __restrict__ se,
    const __bf16* __restrict__ gwp, const float* __restrict__ gb,
    const __bf16* __restrict__ pwp, const float* __restrict__ pb,
    const float* __restrict__ lnw, const float* __restrict__ lnb,
    const __bf16* __restrict__ owp, const float* __restrict__ ob,
    float* __restrict__ hbuf, float* __restrict__ gnsum, float* __restrict__ gnsq) {
    __shared__ __bf16 zt[32 * DCq];   // z_se tile (later reused for normalized h)
    __shared__ __bf16 hb[32 * DCq];   // GLU output h (bf16)
    __bf16* hn = zt;                  // alias: reused after barrier
    int tid = threadIdx.x;
    long long t0 = (long long)blockIdx.x * 32;
    int b = (int)(t0 / Tq);

    for (int i = tid; i < 32 * DCq; i += 256) {
        int m = i / DCq, n = i - m * DCq;
        float zv = (float)zbuf[(t0 + m) * DCq + n];
        zt[i] = (__bf16)(zv * se[b * DCq + n]);
    }
    __syncthreads();

    int wv = tid >> 5, lane = tid & 31, nc = lane & 15, half = lane >> 4;

    // g = sigmoid(z_se @ gw + gb), p = z_se @ pw + pb, h = g*p + (1-g)*z_se
    for (int nt = wv; nt < 40; nt += 8) {
        v8f cg0 = {}, cg1 = {}, cp0 = {}, cp1 = {};
        for (int kt = 0; kt < 20; ++kt) {
            v16bf bg = load_b(gwp + ((size_t)(kt * 40 + nt)) * 512, lane);
            v16bf bp = load_b(pwp + ((size_t)(kt * 40 + nt)) * 512, lane);
            v16bf a0 = load_a(zt,            DCq, kt * 32, lane);
            v16bf a1 = load_a(zt + 16 * DCq, DCq, kt * 32, lane);
            cg0 = bwmma(a0, bg, cg0);
            cg1 = bwmma(a1, bg, cg1);
            cp0 = bwmma(a0, bp, cp0);
            cp1 = bwmma(a1, bp, cp1);
            if (kt + 1 < 20)
                __builtin_prefetch(gwp + ((size_t)((kt + 1) * 40 + nt)) * 512 + lane * 16, 0, 0);
        }
        int n = nt * 16 + nc;
        float gbi = gb[n], pbi = pb[n];
#pragma unroll
        for (int v = 0; v < 8; ++v) {
            int m = v + 8 * half;
            float gv = fast_sigmoid(cg0[v] + gbi);
            float pv = cp0[v] + pbi;
            float zv = (float)zt[m * DCq + n];
            hb[m * DCq + n] = (__bf16)(gv * pv + (1.f - gv) * zv);
            gv = fast_sigmoid(cg1[v] + gbi);
            pv = cp1[v] + pbi;
            zv = (float)zt[(m + 16) * DCq + n];
            hb[(m + 16) * DCq + n] = (__bf16)(gv * pv + (1.f - gv) * zv);
        }
    }
    __syncthreads();

    // LayerNorm over 640 per row; each wave handles 4 of 32 rows
    for (int r = wv; r < 32; r += 8) {
        float s = 0.f, sq = 0.f;
        for (int c = lane; c < DCq; c += 32) { float v = (float)hb[r * DCq + c]; s += v; sq += v * v; }
        s = wred(s); sq = wred(sq);
        float mu = s * (1.f / DCq);
        float var = sq * (1.f / DCq) - mu * mu;
        float rs = fast_rsqrt(var + EPSq);
        for (int c = lane; c < DCq; c += 32) {
            float v = ((float)hb[r * DCq + c] - mu) * rs * lnw[c] + lnb[c];
            hn[r * DCq + c] = (__bf16)v;
        }
    }
    __syncthreads();

    // output projection 640 -> 512 + GroupNorm stats (8 groups of 64 ch)
    for (int nt = wv; nt < 32; nt += 8) {
        v8f c0 = {}, c1 = {};
        for (int kt = 0; kt < 20; ++kt) {
            v16bf bb = load_b(owp + ((size_t)(kt * 32 + nt)) * 512, lane);
            v16bf a0 = load_a(hn,            DCq, kt * 32, lane);
            v16bf a1 = load_a(hn + 16 * DCq, DCq, kt * 32, lane);
            c0 = bwmma(a0, bb, c0);
            c1 = bwmma(a1, bb, c1);
        }
        int n = nt * 16 + nc;
        float bias = ob[n];
        float s = 0.f, sq = 0.f;
#pragma unroll
        for (int v = 0; v < 8; ++v) {
            int m = v + 8 * half;
            float ta = c0[v] + bias;
            float tb = c1[v] + bias;
            hbuf[(t0 + m) * Dq + n]      = ta;
            hbuf[(t0 + m + 16) * Dq + n] = tb;
            s += ta + tb; sq += ta * ta + tb * tb;
        }
        s = wred(s); sq = wred(sq);              // whole 32x16 tile -> one group
        if (lane == 0) {
            int g = n >> 6;
            atomicAdd(&gnsum[b * 8 + g], s);
            atomicAdd(&gnsq[b * 8 + g], sq);
        }
    }
}

// ---------------- K4: GroupNorm finalize ----------------
__global__ __launch_bounds__(256) void k_gn(
    const float* __restrict__ hbuf, const float* __restrict__ gnsum,
    const float* __restrict__ gnsq, const float* __restrict__ gnw,
    const float* __restrict__ gnb, float* __restrict__ out) {
    long long i = (long long)blockIdx.x * 256 + threadIdx.x;
    int b = (int)(i >> 20);                  // T*D = 2^20
    int c = (int)(i & (Dq - 1));
    int g = c >> 6;
    const float inv = 1.f / (float)(Tq * 64);
    float m = gnsum[b * 8 + g] * inv;
    float v = gnsq[b * 8 + g] * inv - m * m;
    float rs = fast_rsqrt(v + EPSq);
    out[i] = (hbuf[i] - m) * rs * gnw[c] + gnb[c];
}

// ---------------- host launch ----------------
extern "C" void kernel_launch(void* const* d_in, const int* in_sizes, int n_in,
                              void* d_out, int out_size, void* d_ws, size_t ws_size,
                              hipStream_t stream) {
    const float* x    = (const float*)d_in[0];
    const float* hw1  = (const float*)d_in[1];
    const float* hb1  = (const float*)d_in[2];
    const float* hw2  = (const float*)d_in[3];
    const float* hb2  = (const float*)d_in[4];
    const float* lw1  = (const float*)d_in[5];
    const float* lb1  = (const float*)d_in[6];
    const float* lw2  = (const float*)d_in[7];
    const float* lb2  = (const float*)d_in[8];
    const float* ew1  = (const float*)d_in[9];
    const float* eb1  = (const float*)d_in[10];
    const float* ew2  = (const float*)d_in[11];
    const float* eb2  = (const float*)d_in[12];
    const float* sew1 = (const float*)d_in[13];
    const float* sew2 = (const float*)d_in[14];
    const float* gw   = (const float*)d_in[15];
    const float* gb   = (const float*)d_in[16];
    const float* pw   = (const float*)d_in[17];
    const float* pb   = (const float*)d_in[18];
    const float* lnw  = (const float*)d_in[19];
    const float* lnb  = (const float*)d_in[20];
    const float* ow   = (const float*)d_in[21];
    const float* ob   = (const float*)d_in[22];
    const float* gnw  = (const float*)d_in[23];
    const float* gnb  = (const float*)d_in[24];

    char* ws = (char*)d_ws;
    __bf16* w1p = (__bf16*)(ws + OFF_W1P);
    __bf16* w2p = (__bf16*)(ws + OFF_W2P);
    __bf16* gwp = (__bf16*)(ws + OFF_GWP);
    __bf16* pwp = (__bf16*)(ws + OFF_PWP);
    __bf16* owp = (__bf16*)(ws + OFF_OWP);
    float*  b1  = (float*)(ws + OFF_B1);
    float*  b2  = (float*)(ws + OFF_B2);
    float*  zsum  = (float*)(ws + OFF_STAT);
    float*  gnsum = zsum + 40960;
    float*  gnsq  = zsum + 41472;
    float*  sebuf = (float*)(ws + OFF_SE);
    __bf16* zbuf  = (__bf16*)(ws + OFF_ZBUF);
    float*  hbuf  = (float*)(ws + OFF_HBUF);
    float*  out   = (float*)d_out;

    // stats must start at zero every call (atomic accumulation)
    k_zero<<<(41984 + 255) / 256, 256, 0, stream>>>(zsum, 41984);

    // weight packing (fp32 -> bf16 fragment tiles)
    k_pack_w1<<<(2 * 40 * 512) / 256, 256, 0, stream>>>(hw1, lw1, ew1, w1p);
    k_pack_w2<<<(20 * 40 * 512) / 256, 256, 0, stream>>>(hw2, lw2, ew2, w2p);
    k_pack<<<(20 * 40 * 512) / 256, 256, 0, stream>>>(gw, DCq, DCq, 20, gwp);
    k_pack<<<(20 * 40 * 512) / 256, 256, 0, stream>>>(pw, DCq, DCq, 20, pwp);
    k_pack<<<(20 * 32 * 512) / 256, 256, 0, stream>>>(ow, DCq, Dq, 20, owp);
    k_pack_bias<<<3, 256, 0, stream>>>(hb1, lb1, eb1, hb2, lb2, eb2, b1, b2);

    // main pipeline (M = 32 token tiles)
    k_proj<<<BT / 32, 256, 0, stream>>>(x, w1p, w2p, b1, b2, zbuf, zsum);
    k_se<<<Bq, 256, 0, stream>>>(zsum, sew1, sew2, sebuf);
    k_gate<<<BT / 32, 256, 0, stream>>>(zbuf, sebuf, gwp, gb, pwp, pb,
                                        lnw, lnb, owp, ob, hbuf, gnsum, gnsq);
    k_gn<<<(int)(((long long)Bq * Tq * Dq) / 256), 256, 0, stream>>>(
        hbuf, gnsum, gnsq, gnw, gnb, out);
}
